// DMSA_72232759984616
// MI455X (gfx1250) — matrix operand order repaired and checked by reading, not compiled, not run on hardware
//
#include <hip/hip_runtime.h>
#include <hip/hip_bf16.h>
#include <math.h>

#define B_ 2
#define H_ 128
#define W_ 128
#define C_ 256
#define HEADS_ 8
#define DH_ 32
#define HD_ 256
#define NPIX (H_ * W_)   /* 16384 */
#define NB (B_ * NPIX)   /* 32768 */

typedef _Float16 h16v __attribute__((ext_vector_type(16)));
typedef float f8v __attribute__((ext_vector_type(8)));

// ----------------------------------------------------------------- TDM path
#if defined(__AMDGCN__) && __has_builtin(__builtin_amdgcn_tensor_load_to_lds)
#define USE_TDM 1
typedef unsigned int u32x4 __attribute__((ext_vector_type(4)));
typedef int i32x4 __attribute__((ext_vector_type(4)));
typedef int i32x8 __attribute__((ext_vector_type(8)));

// Issue one TDM 2D tile load: `tile_rows` rows x 32 f16, row stride
// `stride_elems`; LDS destination padded 4 dwords after every 16 dwords
// (row stride becomes 40 f16 -> bank-conflict-free fragment reads).
// rows_valid / cols_valid bound the tensor for zero-fill OOB handling.
__device__ __forceinline__ void tdm_load_tile_f16(unsigned lds_byte_addr,
                                                  const _Float16* gsrc,
                                                  unsigned tile_rows,
                                                  unsigned rows_valid,
                                                  unsigned cols_valid,
                                                  unsigned stride_elems) {
    unsigned long long ga = (unsigned long long)(size_t)gsrc;
    u32x4 g0 = {
        1u,                                            // count=1, user descriptor
        lds_byte_addr,                                 // lds_addr
        (unsigned)(ga & 0xffffffffull),                // global_addr[31:0]
        (unsigned)((ga >> 32) & 0x01ffffffull) | (2u << 30)  // addr[56:32] | type=2
    };
    unsigned td0 = cols_valid;
    unsigned td1 = rows_valid;
    i32x8 g1;
    // data_size=2B (code 1) | pad_enable | pad_interval=16dw (3) | pad_amount=4dw (3)
    g1[0] = (int)((1u << 16) | (1u << 20) | (3u << 22) | (3u << 25));
    g1[1] = (int)((td0 & 0xffffu) << 16);                          // tensor_dim0[15:0]
    g1[2] = (int)(((td0 >> 16) & 0xffffu) | ((td1 & 0xffffu) << 16));
    g1[3] = (int)(((td1 >> 16) & 0xffffu) | (32u << 16));          // tile_dim0=32
    g1[4] = (int)(tile_rows & 0xffffu);                            // tile_dim1, tile_dim2=0
    g1[5] = (int)stride_elems;                                     // tensor_dim0_stride[31:0]
    g1[6] = 0;
    g1[7] = 0;
    i32x4 z4 = {0, 0, 0, 0};
#if __clang_major__ >= 23
    i32x8 z8 = {0, 0, 0, 0, 0, 0, 0, 0};
    __builtin_amdgcn_tensor_load_to_lds(g0, g1, z4, z4, z8, 0);
#else
    __builtin_amdgcn_tensor_load_to_lds(g0, g1, z4, z4, 0);
#endif
}
#endif

// ---------------------------------------------------------------- converts
__global__ __launch_bounds__(256)
void cvt_f32_f16_kernel(const float* __restrict__ s, _Float16* __restrict__ d, int n) {
    int i = blockIdx.x * 256 + threadIdx.x;
    if (i < n) d[i] = (_Float16)s[i];
}

// build xy = concat(x, y) along channel, f16: (NB, 512)
__global__ __launch_bounds__(256)
void pack_xy_kernel(const float* __restrict__ x, const float* __restrict__ y,
                    _Float16* __restrict__ xy) {
    int i = blockIdx.x * 256 + threadIdx.x;   // over NB*512
    int col = i & 511;
    int row = i >> 9;
    float v = (col < C_) ? x[(size_t)row * C_ + col] : y[(size_t)row * C_ + (col - C_)];
    xy[i] = (_Float16)v;
}

// ---------------------------------------------------------------- WMMA GEMM
// Out[m, o] = act( sum_k A[m*lda + k] * Wt[o*ldw + k] + bias[o] )
// 64x64 block tile, 4 waves (128 thr), each wave 2x2 WMMA accumulators.
// Tiles streamed by the Tensor Data Mover (double buffered) when available.
template <int ACT, int OUT16>
__global__ __launch_bounds__(128)
void gemm_wmma_kernel(const _Float16* __restrict__ A, int lda,
                      const _Float16* __restrict__ Wt, int ldw,
                      const float* __restrict__ bias,
                      void* __restrict__ Out, int ldo,
                      int M, int K, int O) {
    const int tid  = threadIdx.x;
    const int lane = tid & 31;
    const int wave = tid >> 5;
    const int wm = wave >> 1, wn = wave & 1;
    const int m0 = blockIdx.y * 64;
    const int o0 = blockIdx.x * 64;

    f8v acc[2][2] = {};

    const int lrow  = lane & 15;
    const int ahalf = (lane >> 4) << 3;   // 0 / 8  (A K-offset per ISA layout)
    const int bhalf = (lane >> 4) << 4;   // 0 / 16 (unused: B uses ahalf, same k order as A)

    // one K-step of WMMA work from padded LDS tiles (row stride 40 f16)
    auto compute_step = [&](const _Float16 (*Ab)[40], const _Float16 (*Wb)[40]) {
        h16v afr[2], bfr[2];
#pragma unroll
        for (int ti = 0; ti < 2; ++ti) {
            int r = wm * 32 + ti * 16 + lrow;
            h16v a;
#pragma unroll
            for (int j = 0; j < 8; ++j) {
                int kk = ((j >> 2) << 4) + ahalf + ((j & 3) << 1);
                a[2 * j]     = Ab[r][kk];
                a[2 * j + 1] = Ab[r][kk + 1];
            }
            afr[ti] = a;
        }
#pragma unroll
        for (int tj = 0; tj < 2; ++tj) {
            int ci = wn * 32 + tj * 16 + lrow;
            h16v bb;
#pragma unroll
            for (int j = 0; j < 8; ++j) {
                int kk = ((j >> 2) << 4) + ahalf + ((j & 3) << 1);
                bb[2 * j]     = Wb[ci][kk];
                bb[2 * j + 1] = Wb[ci][kk + 1];
            }
            bfr[tj] = bb;
        }
#pragma unroll
        for (int ti = 0; ti < 2; ++ti)
#pragma unroll
            for (int tj = 0; tj < 2; ++tj)
                acc[ti][tj] = __builtin_amdgcn_wmma_f32_16x16x32_f16(
                    false, afr[ti], false, bfr[tj], (short)0, acc[ti][tj], false, false);
    };

#ifdef USE_TDM
    __shared__ _Float16 As[2][64][40];
    __shared__ _Float16 Ws[2][64][40];
    const int ntiles = K >> 5;
    const unsigned arows = (unsigned)(M - m0);
    const unsigned wrows = (unsigned)(O > o0 ? O - o0 : 0);
    if (wave == 0) {   // wave-level TDM issue: tile 0 pair
        tdm_load_tile_f16((unsigned)(size_t)&As[0][0][0],
                          A + (size_t)m0 * lda, 64u, arows, (unsigned)K, (unsigned)lda);
        tdm_load_tile_f16((unsigned)(size_t)&Ws[0][0][0],
                          Wt + (size_t)o0 * ldw, 64u, wrows, (unsigned)K, (unsigned)ldw);
    }
    for (int t = 0; t < ntiles; ++t) {
        const int buf = t & 1;
        if (wave == 0) {
            if (t + 1 < ntiles) {   // stream next tile pair into the other buffer
                int k0n = (t + 1) << 5;
                tdm_load_tile_f16((unsigned)(size_t)&As[buf ^ 1][0][0],
                                  A + (size_t)m0 * lda + k0n, 64u, arows,
                                  (unsigned)(K - k0n), (unsigned)lda);
                tdm_load_tile_f16((unsigned)(size_t)&Ws[buf ^ 1][0][0],
                                  Wt + (size_t)o0 * ldw + k0n, 64u, wrows,
                                  (unsigned)(K - k0n), (unsigned)ldw);
                __builtin_amdgcn_s_wait_tensorcnt(2);  // tile t pair complete
            } else {
                __builtin_amdgcn_s_wait_tensorcnt(0);
            }
        }
        __syncthreads();
        compute_step(As[buf], Ws[buf]);
        __syncthreads();
    }
#else
    __shared__ _Float16 As[64][40];
    __shared__ _Float16 Ws[64][40];
    for (int k0 = 0; k0 < K; k0 += 32) {
#pragma unroll
        for (int it = 0; it < 2; ++it) {
            int chunk = tid + it * 128;           // 0..255
            int row = chunk >> 2;
            int cc  = (chunk & 3) << 3;           // 0,8,16,24
            uint4 av; av.x = av.y = av.z = av.w = 0u;
            int am = m0 + row;
            if (am < M) av = *(const uint4*)(A + (size_t)am * lda + k0 + cc);
            *(uint4*)(&As[row][cc]) = av;
            uint4 wv; wv.x = wv.y = wv.z = wv.w = 0u;
            int wo = o0 + row;
            if (wo < O) wv = *(const uint4*)(Wt + (size_t)wo * ldw + k0 + cc);
            *(uint4*)(&Ws[row][cc]) = wv;
        }
        if (k0 + 32 < K)
            __builtin_prefetch(A + (size_t)(m0 + (tid >> 2)) * lda + k0 + 32, 0, 0);
        __syncthreads();
        compute_step(As, Ws);
        __syncthreads();
    }
#endif

    const int mh = (lane >> 4) << 3;  // C/D: lanes 16..31 hold M = r + 8
#pragma unroll
    for (int ti = 0; ti < 2; ++ti)
#pragma unroll
        for (int tj = 0; tj < 2; ++tj)
#pragma unroll
            for (int r = 0; r < 8; ++r) {
                int m = m0 + wm * 32 + ti * 16 + r + mh;
                int o = o0 + wn * 32 + tj * 16 + lrow;
                if (m < M && o < O) {
                    float v = acc[ti][tj][r];
                    if (bias) v += bias[o];
                    if (ACT == 1) v = v > 0.f ? v : 0.01f * v;
                    if (OUT16) ((_Float16*)Out)[(size_t)m * ldo + o] = (_Float16)v;
                    else       ((float*)Out)[(size_t)m * ldo + o] = v;
                }
            }
}

// ---------------------------------------------------------------- row L2 inv-norms
// t: (B, NPIX, HD) f16; invn[b*HD + c] = 1/max(||t[b,:,c]||, 1e-12)
__global__ __launch_bounds__(256)
void rownorm_kernel(const _Float16* __restrict__ t, float* __restrict__ invn) {
    __shared__ float red[256];
    int cidx = blockIdx.x;
    int b = cidx >> 8;
    int c = cidx & 255;
    const _Float16* p = t + (size_t)b * NPIX * HD_ + c;
    float s = 0.f;
    for (int n = threadIdx.x; n < NPIX; n += 256) {
        float v = (float)p[(size_t)n * HD_];
        s += v * v;
    }
    red[threadIdx.x] = s;
    __syncthreads();
    for (int off = 128; off > 0; off >>= 1) {
        if (threadIdx.x < off) red[threadIdx.x] += red[threadIdx.x + off];
        __syncthreads();
    }
    if (threadIdx.x == 0) invn[cidx] = 1.f / fmaxf(sqrtf(red[0]), 1e-12f);
}

// ---------------------------------------------------------------- attention logits
// logits[idx, d, e] = rescale * invk[d] * invq[e] * sum_n K[n,d] * Q[n,e]
// idx = dir*16 + b*8 + h ; one wave per (dir,b,h); WMMA over K=NPIX.
// Tiles staged UN-transposed ([n][d] / [n][e]): B reads Q's natural layout,
// A reads K column-wise -- no LDS scatter needed. TDM-streamed, double buffer,
// no barriers (single wave, TENSORcnt orders TDM vs ds reads).
__global__ __launch_bounds__(32)
void attn_logits_kernel(const _Float16* __restrict__ kx, const _Float16* __restrict__ ky,
                        const _Float16* __restrict__ qx, const _Float16* __restrict__ qy,
                        const float* __restrict__ rescale_x, const float* __restrict__ rescale_y,
                        const float* __restrict__ inv_qx, const float* __restrict__ inv_kx,
                        const float* __restrict__ inv_qy, const float* __restrict__ inv_ky,
                        float* __restrict__ logits) {
    int idx = blockIdx.x;
    int dir = idx >> 4;
    int b = (idx >> 3) & 1;
    int h = idx & 7;
    const _Float16* kt = dir ? kx : ky;
    const _Float16* qt = dir ? qy : qx;
    const float* invk  = dir ? inv_kx : inv_ky;
    const float* invq  = dir ? inv_qy : inv_qx;
    float resc = dir ? rescale_y[h] : rescale_x[h];
    const _Float16* kp = kt + (size_t)b * NPIX * HD_ + h * DH_;
    const _Float16* qp = qt + (size_t)b * NPIX * HD_ + h * DH_;

    const int lane  = threadIdx.x;
    const int lrow  = lane & 15;
    const int ahalf = (lane >> 4) << 3;
    const int bhalf = (lane >> 4) << 4;
    f8v acc[2][2] = {};

    // Fragment build from [n][d]-layout tiles (row stride 40 f16):
    //   A[d_row][kk] = Kt[kk][d_row]   (column walk, banks 20*kk mod 64: conflict-free)
    //   B[kk][e]     = Qt[kk][e]       (natural layout)
    auto compute_step = [&](const _Float16 (*Kt)[40], const _Float16 (*Qt)[40]) {
        h16v afr[2], bfr[2];
#pragma unroll
        for (int ti = 0; ti < 2; ++ti) {
            int d = ti * 16 + lrow;
            h16v a;
#pragma unroll
            for (int j = 0; j < 8; ++j) {
                int kk = ((j >> 2) << 4) + ahalf + ((j & 3) << 1);
                a[2 * j]     = Kt[kk][d];
                a[2 * j + 1] = Kt[kk + 1][d];
            }
            afr[ti] = a;
        }
#pragma unroll
        for (int tj = 0; tj < 2; ++tj) {
            int e = tj * 16 + lrow;
            h16v bb;
#pragma unroll
            for (int j = 0; j < 8; ++j) {
                int kk = ((j >> 2) << 4) + ahalf + ((j & 3) << 1);
                bb[2 * j]     = Qt[kk][e];
                bb[2 * j + 1] = Qt[kk + 1][e];
            }
            bfr[tj] = bb;
        }
#pragma unroll
        for (int ti = 0; ti < 2; ++ti)
#pragma unroll
            for (int tj = 0; tj < 2; ++tj)
                acc[ti][tj] = __builtin_amdgcn_wmma_f32_16x16x32_f16(
                    false, afr[ti], false, bfr[tj], (short)0, acc[ti][tj], false, false);
    };

#ifdef USE_TDM
    __shared__ _Float16 Kt[2][32][40];
    __shared__ _Float16 Qt[2][32][40];
    const int ntiles = NPIX >> 5;
    tdm_load_tile_f16((unsigned)(size_t)&Kt[0][0][0], kp, 32u, (unsigned)NPIX, 32u, HD_);
    tdm_load_tile_f16((unsigned)(size_t)&Qt[0][0][0], qp, 32u, (unsigned)NPIX, 32u, HD_);
    for (int t = 0; t < ntiles; ++t) {
        const int buf = t & 1;
        if (t + 1 < ntiles) {
            const _Float16* kn = kp + (size_t)((t + 1) << 5) * HD_;
            const _Float16* qn = qp + (size_t)((t + 1) << 5) * HD_;
            tdm_load_tile_f16((unsigned)(size_t)&Kt[buf ^ 1][0][0], kn, 32u,
                              (unsigned)(NPIX - ((t + 1) << 5)), 32u, HD_);
            tdm_load_tile_f16((unsigned)(size_t)&Qt[buf ^ 1][0][0], qn, 32u,
                              (unsigned)(NPIX - ((t + 1) << 5)), 32u, HD_);
            __builtin_amdgcn_s_wait_tensorcnt(2);
        } else {
            __builtin_amdgcn_s_wait_tensorcnt(0);
        }
        compute_step(Kt[buf], Qt[buf]);
    }
#else
    __shared__ _Float16 Kt[32][40];
    __shared__ _Float16 Qt[32][40];
    for (int n0 = 0; n0 < NPIX; n0 += 32) {
        const _Float16* kc = kp + (size_t)(n0 + lane) * HD_;
        const _Float16* qc = qp + (size_t)(n0 + lane) * HD_;
#pragma unroll
        for (int cc = 0; cc < 32; cc += 8) {   // lane copies its own row, vectorized
            *(uint4*)(&Kt[lane][cc]) = *(const uint4*)(kc + cc);
            *(uint4*)(&Qt[lane][cc]) = *(const uint4*)(qc + cc);
        }
        __syncthreads();
        compute_step(Kt, Qt);
        __syncthreads();
    }
#endif

    const int mh = (lane >> 4) << 3;
#pragma unroll
    for (int ti = 0; ti < 2; ++ti)
#pragma unroll
        for (int tj = 0; tj < 2; ++tj)
#pragma unroll
            for (int r = 0; r < 8; ++r) {
                int d = ti * 16 + r + mh;
                int e = tj * 16 + lrow;
                float v = acc[ti][tj][r] * resc *
                          invk[b * HD_ + h * DH_ + d] * invq[b * HD_ + h * DH_ + e];
                logits[(size_t)idx * 1024 + d * 32 + e] = v;
            }
}

// ---------------------------------------------------------------- softmax (rows of 32)
__global__ __launch_bounds__(32)
void softmax_kernel(const float* __restrict__ logits, _Float16* __restrict__ attn) {
    int row = blockIdx.x * 32 + threadIdx.x;   // 1024 rows total
    const float* p = logits + (size_t)row * 32;
    float mx = -1e30f;
    for (int e = 0; e < 32; ++e) mx = fmaxf(mx, p[e]);
    float ex[32], s = 0.f;
    for (int e = 0; e < 32; ++e) { ex[e] = expf(p[e] - mx); s += ex[e]; }
    float inv = 1.f / s;
    _Float16* o = attn + (size_t)row * 32;
    for (int e = 0; e < 32; ++e) o[e] = (_Float16)(ex[e] * inv);
}

// ---------------------------------------------------------------- depthwise 3x3 + GELU(exact)
__global__ __launch_bounds__(256)
void dwconv_gelu_kernel(const _Float16* __restrict__ v, const float* __restrict__ w,
                        const float* __restrict__ bias, float* __restrict__ tmp) {
    int i = blockIdx.x * 256 + threadIdx.x;   // over NB*C
    int c = i & 255;
    int x = (i >> 8) & 127;
    int y = (i >> 15) & 127;
    int b = i >> 22;
    float s = bias[c];
#pragma unroll
    for (int dy = 0; dy < 3; ++dy) {
        int yy = y + dy - 1;
        if (yy < 0 || yy >= H_) continue;
#pragma unroll
        for (int dx = 0; dx < 3; ++dx) {
            int xx = x + dx - 1;
            if (xx < 0 || xx >= W_) continue;
            s += (float)v[(((size_t)b * H_ + yy) * W_ + xx) * C_ + c] * w[c * 9 + dy * 3 + dx];
        }
    }
    tmp[i] = 0.5f * s * (1.f + erff(s * 0.70710678118654752f));
}

// depthwise 3x3 + accumulate into out
__global__ __launch_bounds__(256)
void dwconv_add_kernel(const float* __restrict__ t, const float* __restrict__ w,
                       const float* __restrict__ bias, float* __restrict__ out) {
    int i = blockIdx.x * 256 + threadIdx.x;
    int c = i & 255;
    int x = (i >> 8) & 127;
    int y = (i >> 15) & 127;
    int b = i >> 22;
    float s = bias[c];
#pragma unroll
    for (int dy = 0; dy < 3; ++dy) {
        int yy = y + dy - 1;
        if (yy < 0 || yy >= H_) continue;
#pragma unroll
        for (int dx = 0; dx < 3; ++dx) {
            int xx = x + dx - 1;
            if (xx < 0 || xx >= W_) continue;
            s += t[(((size_t)b * H_ + yy) * W_ + xx) * C_ + c] * w[c * 9 + dy * 3 + dx];
        }
    }
    out[i] += s;
}

// ---------------------------------------------------------------- host launch
extern "C" void kernel_launch(void* const* d_in, const int* in_sizes, int n_in,
                              void* d_out, int out_size, void* d_ws, size_t ws_size,
                              hipStream_t stream) {
    (void)in_sizes; (void)n_in; (void)out_size; (void)ws_size;
    const float* x_in = (const float*)d_in[0];
    const float* y_in = (const float*)d_in[1];
    const float* fx_w1 = (const float*)d_in[2];
    const float* fx_b1 = (const float*)d_in[3];
    const float* fx_w2 = (const float*)d_in[4];
    const float* fx_b2 = (const float*)d_in[5];
    const float* fy_w1 = (const float*)d_in[6];
    const float* fy_b1 = (const float*)d_in[7];
    const float* fy_w2 = (const float*)d_in[8];
    const float* fy_b2 = (const float*)d_in[9];
    const float* q_w1 = (const float*)d_in[10];
    const float* q_b1 = (const float*)d_in[11];
    const float* q_w2 = (const float*)d_in[12];
    const float* k_w1 = (const float*)d_in[13];
    const float* k_b1 = (const float*)d_in[14];
    const float* k_w2 = (const float*)d_in[15];
    const float* v_w1 = (const float*)d_in[16];
    const float* v_b1 = (const float*)d_in[17];
    const float* v_w2 = (const float*)d_in[18];
    const float* rescale_x = (const float*)d_in[19];
    const float* rescale_y = (const float*)d_in[20];
    const float* px_w = (const float*)d_in[21];
    const float* px_b = (const float*)d_in[22];
    const float* py_w = (const float*)d_in[23];
    const float* py_b = (const float*)d_in[24];
    const float* pe_w1 = (const float*)d_in[25];
    const float* pe_b1 = (const float*)d_in[26];
    const float* pe_w2 = (const float*)d_in[27];
    const float* pe_b2 = (const float*)d_in[28];

    char* ws = (char*)d_ws;
    size_t off = 0;
    auto alloc = [&](size_t bytes) -> void* {
        void* p = ws + off;
        off = (off + bytes + 255) & ~(size_t)255;
        return p;
    };

    _Float16* xy  = (_Float16*)alloc((size_t)NB * 512 * 2);
    _Float16* hid = (_Float16*)alloc((size_t)NB * 256 * 2);
    _Float16* fkx = (_Float16*)alloc((size_t)NB * 256 * 2);
    _Float16* fky = (_Float16*)alloc((size_t)NB * 256 * 2);
    _Float16* qxb = (_Float16*)alloc((size_t)NB * 256 * 2);
    _Float16* kxb = (_Float16*)alloc((size_t)NB * 256 * 2);
    _Float16* vxb = (_Float16*)alloc((size_t)NB * 256 * 2);
    _Float16* qyb = (_Float16*)alloc((size_t)NB * 256 * 2);
    _Float16* kyb = (_Float16*)alloc((size_t)NB * 256 * 2);
    _Float16* vyb = (_Float16*)alloc((size_t)NB * 256 * 2);
    _Float16* xo  = (_Float16*)alloc((size_t)NB * 256 * 2);
    _Float16* yo  = (_Float16*)alloc((size_t)NB * 256 * 2);
    _Float16* wfx1 = (_Float16*)alloc(131072 * 2);
    _Float16* wfx2 = (_Float16*)alloc(65536 * 2);
    _Float16* wfy1 = (_Float16*)alloc(131072 * 2);
    _Float16* wfy2 = (_Float16*)alloc(65536 * 2);
    _Float16* wq1  = (_Float16*)alloc(65536 * 2);
    _Float16* wq2  = (_Float16*)alloc(65536 * 2);
    _Float16* wk1  = (_Float16*)alloc(65536 * 2);
    _Float16* wk2  = (_Float16*)alloc(65536 * 2);
    _Float16* wv1  = (_Float16*)alloc(65536 * 2);
    _Float16* wv2  = (_Float16*)alloc(65536 * 2);
    _Float16* wpx  = (_Float16*)alloc(65536 * 2);
    _Float16* wpy  = (_Float16*)alloc(65536 * 2);
    float* invqx = (float*)alloc(B_ * HD_ * 4);
    float* invkx = (float*)alloc(B_ * HD_ * 4);
    float* invqy = (float*)alloc(B_ * HD_ * 4);
    float* invky = (float*)alloc(B_ * HD_ * 4);
    float* logits = (float*)alloc(32 * 1024 * 4);
    _Float16* attn = (_Float16*)alloc(32 * 1024 * 2);
    float* tmp = (float*)alloc((size_t)NB * C_ * 4);

    auto cvt = [&](const float* s, _Float16* d, int n) {
        cvt_f32_f16_kernel<<<(n + 255) / 256, 256, 0, stream>>>(s, d, n);
    };
    cvt(fx_w1, wfx1, 131072); cvt(fx_w2, wfx2, 65536);
    cvt(fy_w1, wfy1, 131072); cvt(fy_w2, wfy2, 65536);
    cvt(q_w1, wq1, 65536);    cvt(q_w2, wq2, 65536);
    cvt(k_w1, wk1, 65536);    cvt(k_w2, wk2, 65536);
    cvt(v_w1, wv1, 65536);    cvt(v_w2, wv2, 65536);
    cvt(px_w, wpx, 65536);    cvt(py_w, wpy, 65536);

    pack_xy_kernel<<<(NB * 512) / 256, 256, 0, stream>>>(x_in, y_in, xy);

    auto gemm = [&](int act, int out16, const _Float16* A, int lda,
                    const _Float16* Wt, int ldw, const float* bias,
                    void* Out, int ldo, int M, int K, int O) {
        dim3 g((O + 63) / 64, (M + 63) / 64);
        if (act == 1)
            gemm_wmma_kernel<1, 1><<<g, 128, 0, stream>>>(A, lda, Wt, ldw, bias, Out, ldo, M, K, O);
        else if (out16)
            gemm_wmma_kernel<0, 1><<<g, 128, 0, stream>>>(A, lda, Wt, ldw, bias, Out, ldo, M, K, O);
        else
            gemm_wmma_kernel<0, 0><<<g, 128, 0, stream>>>(A, lda, Wt, ldw, bias, Out, ldo, M, K, O);
    };

    // fusion_x / fusion_y
    gemm(1, 1, xy, 512, wfx1, 512, fx_b1, hid, 256, NB, 512, 256);
    gemm(0, 1, hid, 256, wfx2, 256, fx_b2, fkx, 256, NB, 256, 256);
    gemm(1, 1, xy, 512, wfy1, 512, fy_b1, hid, 256, NB, 512, 256);
    gemm(0, 1, hid, 256, wfy2, 256, fy_b2, fky, 256, NB, 256, 256);
    // q/v from x,y (views into xy); k from fused features
    gemm(1, 1, xy, 512, wq1, 256, q_b1, hid, 256, NB, 256, 256);
    gemm(0, 1, hid, 256, wq2, 256, nullptr, qxb, 256, NB, 256, 256);
    gemm(1, 1, xy + 256, 512, wq1, 256, q_b1, hid, 256, NB, 256, 256);
    gemm(0, 1, hid, 256, wq2, 256, nullptr, qyb, 256, NB, 256, 256);
    gemm(1, 1, fkx, 256, wk1, 256, k_b1, hid, 256, NB, 256, 256);
    gemm(0, 1, hid, 256, wk2, 256, nullptr, kxb, 256, NB, 256, 256);
    gemm(1, 1, fky, 256, wk1, 256, k_b1, hid, 256, NB, 256, 256);
    gemm(0, 1, hid, 256, wk2, 256, nullptr, kyb, 256, NB, 256, 256);
    gemm(1, 1, xy, 512, wv1, 256, v_b1, hid, 256, NB, 256, 256);
    gemm(0, 1, hid, 256, wv2, 256, nullptr, vxb, 256, NB, 256, 256);
    gemm(1, 1, xy + 256, 512, wv1, 256, v_b1, hid, 256, NB, 256, 256);
    gemm(0, 1, hid, 256, wv2, 256, nullptr, vyb, 256, NB, 256, 256);

    rownorm_kernel<<<B_ * HD_, 256, 0, stream>>>(qxb, invqx);
    rownorm_kernel<<<B_ * HD_, 256, 0, stream>>>(kxb, invkx);
    rownorm_kernel<<<B_ * HD_, 256, 0, stream>>>(qyb, invqy);
    rownorm_kernel<<<B_ * HD_, 256, 0, stream>>>(kyb, invky);

    attn_logits_kernel<<<32, 32, 0, stream>>>(kxb, kyb, qxb, qyb, rescale_x, rescale_y,
                                              invqx, invkx, invqy, invky, logits);
    softmax_kernel<<<32, 32, 0, stream>>>(logits, attn);

    // out[n, d] = sum_e V[n, e] * attn[d, e]  per (dir, b, head)
    for (int dir = 0; dir < 2; ++dir)
        for (int b = 0; b < 2; ++b)
            for (int h = 0; h < 8; ++h) {
                int idx = dir * 16 + b * 8 + h;
                const _Float16* vsrc = (dir ? vyb : vxb) + (size_t)b * NPIX * HD_ + h * DH_;
                _Float16* dst = (dir ? yo : xo) + (size_t)b * NPIX * HD_ + h * DH_;
                gemm(0, 1, vsrc, HD_, attn + (size_t)idx * 1024, 32, nullptr,
                     dst, HD_, NPIX, 32, 32);
            }

    float* outx = (float*)d_out;
    float* outy = outx + (size_t)NB * C_;
    gemm(0, 0, xo, HD_, wpx, HD_, px_b, outx, C_, NB, HD_, C_);
    gemm(0, 0, yo, HD_, wpy, HD_, py_b, outy, C_, NB, HD_, C_);

    int nconv = NB * C_;
    dwconv_gelu_kernel<<<nconv / 256, 256, 0, stream>>>(vxb, pe_w1, pe_b1, tmp);
    dwconv_add_kernel<<<nconv / 256, 256, 0, stream>>>(tmp, pe_w2, pe_b2, outx);
    dwconv_gelu_kernel<<<nconv / 256, 256, 0, stream>>>(vyb, pe_w1, pe_b1, tmp);
    dwconv_add_kernel<<<nconv / 256, 256, 0, stream>>>(tmp, pe_w2, pe_b2, outy);
}
